// MultiHeadedRnn_85023172592671
// MI455X (gfx1250) — matrix operand order repaired and checked by reading, not compiled, not run on hardware
//
#include <hip/hip_runtime.h>
#include <hip/hip_bf16.h>
#include <cstdint>
#include <cstddef>

// Problem dims (fixed by reference)
#define B_DIM 64
#define A_DIM 50
#define V_DIM 256
#define H_DIM 1024
#define T_STEPS 256

typedef __attribute__((ext_vector_type(16))) __bf16 v16bf;
typedef __attribute__((ext_vector_type(8)))  float  v8f;

union FragBF {
  v16bf v;
  uint4 u[2];
};

__device__ __forceinline__ unsigned short f2bf(float f) {
  unsigned int u = __float_as_uint(f);
  return (unsigned short)((u + 0x7FFFu + ((u >> 16) & 1u)) >> 16);  // RNE
}

__device__ __forceinline__ float sigmoidf_(float x) {
  return 1.0f / (1.0f + __expf(-x));
}

// ---------------------------------------------------------------------------
// f32 -> bf16 elementwise conversion (weights / x / h-init), grid-stride.
// ---------------------------------------------------------------------------
__global__ void convert_f32_bf16_kernel(const float* __restrict__ in,
                                        unsigned short* __restrict__ out, long n) {
  long i = (long)blockIdx.x * blockDim.x + threadIdx.x;
  long stride = (long)gridDim.x * blockDim.x;
  for (; i < n; i += stride) out[i] = f2bf(in[i]);
}

// ---------------------------------------------------------------------------
// One LSTM timestep for one layer.
//   gates[64,4096] = Ax[64,Kx]*Wx^T + Ah[64,1024]*Wh^T + b_ih + b_hh
// Workgroup = 256 threads = 8 waves = {4 gates} x {2 K-halves}, one 16x16
// (m,n) output tile per workgroup.  Each wave runs ~20 WMMA K-iterations on
// its (gate, K-half) slice, writes the partial f32 tile to LDS, then all 256
// threads reduce + apply the LSTM pointwise update (1 element each).
// Grid: 256 blocks (4 Mtiles x 64 Ntiles) -> 2048 waves/step.
// ---------------------------------------------------------------------------
__launch_bounds__(256)
__global__ void lstm_step_kernel(
    const unsigned short* __restrict__ Ax, long ldax, int Kx,   // bf16 [64,Kx]
    const unsigned short* __restrict__ Wx,                      // bf16 [4H,Kx] row-major
    const unsigned short* __restrict__ Ah,                      // bf16 [64,1024]
    const unsigned short* __restrict__ Wh,                      // bf16 [4H,1024] row-major
    const float* __restrict__ bih, const float* __restrict__ bhh,
    float* __restrict__ c_state,                                // f32 [64,1024]
    float* __restrict__ h_state,                                // f32 [64,1024]
    unsigned short* __restrict__ h_out_bf)                      // bf16 [64,1024]
{
  __shared__ float gsm[2][4][16][16];   // [khalf][gate][m][n], 8 KB

  const int wave  = threadIdx.x >> 5;   // 0..7
  const int lane  = threadIdx.x & 31;
  const int gate  = wave & 3;           // i,f,g,o
  const int khalf = wave >> 2;          // 0,1
  const int mtile = blockIdx.x & 3;
  const int ntile = blockIdx.x >> 2;    // 0..63
  const int row0  = mtile << 4;
  const int nbase = ntile << 4;

  // A-fragment addressing (16x32 bf16): lane row = lane%16, k-block by lane/16
  const int arow = lane & 15;
  const int akb  = (lane >> 4) << 3;    // 0 or 8
  // B-fragment addressing (32x16 bf16): lane column = lane%16, k-block by lane/16
  const int bn   = lane & 15;
  const int bko  = (lane >> 4) << 3;    // 0 or 8

  v8f acc = {};
  const long jrow = (long)(gate * H_DIM + nbase + bn);

  // ---- x-part slice: K in [khalf*Kx/2, (khalf+1)*Kx/2) ----
  {
    const int kxh = Kx >> 1;
    const int kbeg = khalf * kxh;
    const unsigned short* ap  = Ax + (long)(row0 + arow) * ldax + akb;
    const unsigned short* wp0 = Wx + jrow * Kx + bko;
    for (int k0 = kbeg; k0 < kbeg + kxh; k0 += 32) {
      FragBF a, b;
      a.u[0] = *(const uint4*)(ap + k0);
      a.u[1] = *(const uint4*)(ap + k0 + 16);
      b.u[0] = ((const uint4*)(wp0 + k0))[0];
      b.u[1] = ((const uint4*)(wp0 + k0 + 16))[0];
      acc = __builtin_amdgcn_wmma_f32_16x16x32_bf16(
          false, a.v, false, b.v, (short)0, acc, false, false);
    }
  }

  // ---- h-part slice: K in [khalf*512, khalf*512+512) ----
  {
    const int kbeg = khalf * (H_DIM >> 1);
    const unsigned short* ap  = Ah + (long)(row0 + arow) * H_DIM + akb;
    const unsigned short* wp0 = Wh + jrow * H_DIM + bko;
    for (int k0 = kbeg; k0 < kbeg + (H_DIM >> 1); k0 += 32) {
      FragBF a, b;
      a.u[0] = *(const uint4*)(ap + k0);
      a.u[1] = *(const uint4*)(ap + k0 + 16);
      b.u[0] = ((const uint4*)(wp0 + k0))[0];
      b.u[1] = ((const uint4*)(wp0 + k0 + 16))[0];
      acc = __builtin_amdgcn_wmma_f32_16x16x32_bf16(
          false, a.v, false, b.v, (short)0, acc, false, false);
    }
  }

  // ---- stash partial gate tile (C layout: m = 8*(lane/16)+i, n = lane%16) ----
  {
    const int n  = lane & 15;
    const int mh = (lane >> 4) << 3;
#pragma unroll
    for (int i = 0; i < 8; ++i) gsm[khalf][gate][mh + i][n] = acc[i];
  }
  __syncthreads();

  // ---- pointwise LSTM update: 256 threads x 1 element ----
  {
    const int e   = threadIdx.x;
    const int m   = e >> 4;
    const int n   = e & 15;
    const int col = nbase + n;
    const long idx = (long)(row0 + m) * H_DIM + col;

    const float gi = gsm[0][0][m][n] + gsm[1][0][m][n] + bih[col] + bhh[col];
    const float gf = gsm[0][1][m][n] + gsm[1][1][m][n] + bih[H_DIM + col] + bhh[H_DIM + col];
    const float gg = gsm[0][2][m][n] + gsm[1][2][m][n] + bih[2 * H_DIM + col] + bhh[2 * H_DIM + col];
    const float go = gsm[0][3][m][n] + gsm[1][3][m][n] + bih[3 * H_DIM + col] + bhh[3 * H_DIM + col];

    const float c_new = sigmoidf_(gf) * c_state[idx] + sigmoidf_(gi) * tanhf(gg);
    const float h_new = sigmoidf_(go) * tanhf(c_new);
    c_state[idx]  = c_new;
    h_state[idx]  = h_new;
    h_out_bf[idx] = f2bf(h_new);
  }
}

// ---------------------------------------------------------------------------
// Heads: outs[a,b,v] = last[b,:] . head_w[a,v,:] + head_b[a,v]
// 50 authors x (4 Mtiles x 16 Vtiles) = 3200 waves = 400 blocks x 8 waves.
// ---------------------------------------------------------------------------
__launch_bounds__(256)
__global__ void head_kernel(
    const unsigned short* __restrict__ hlast,   // bf16 [64,1024]
    const unsigned short* __restrict__ hw,      // bf16 [A*V,1024] row-major
    const float* __restrict__ hb,               // f32  [A,V]
    float* __restrict__ out)                    // f32  [A,B,V]
{
  const int wid   = blockIdx.x * 8 + (threadIdx.x >> 5);
  const int lane  = threadIdx.x & 31;
  const int a     = wid >> 6;          // 0..49
  const int rem   = wid & 63;
  const int mtile = rem & 3;
  const int vtile = rem >> 2;          // 0..15
  const int row0  = mtile << 4;
  const int nbase = vtile << 4;

  const int arow = lane & 15;
  const int akb  = (lane >> 4) << 3;
  const int bn   = lane & 15;
  const int bko  = (lane >> 4) << 3;

  v8f acc = {};
  const unsigned short* ap = hlast + (long)(row0 + arow) * H_DIM + akb;
  const unsigned short* wp = hw + (long)(a * V_DIM + nbase + bn) * H_DIM + bko;
  for (int k0 = 0; k0 < H_DIM; k0 += 32) {
    FragBF afr, bfr;
    afr.u[0] = *(const uint4*)(ap + k0);
    afr.u[1] = *(const uint4*)(ap + k0 + 16);
    bfr.u[0] = ((const uint4*)(wp + k0))[0];
    bfr.u[1] = ((const uint4*)(wp + k0 + 16))[0];
    acc = __builtin_amdgcn_wmma_f32_16x16x32_bf16(
        false, afr.v, false, bfr.v, (short)0, acc, false, false);
  }

  const int col  = nbase + (lane & 15);
  const int half = lane >> 4;
  const float bias = hb[a * V_DIM + col];
#pragma unroll
  for (int i = 0; i < 8; ++i) {
    const int m = row0 + half * 8 + i;
    out[(long)a * B_DIM * V_DIM + (long)m * V_DIM + col] = acc[i] + bias;
  }
}

// ---------------------------------------------------------------------------
extern "C" void kernel_launch(void* const* d_in, const int* in_sizes, int n_in,
                              void* d_out, int out_size, void* d_ws, size_t ws_size,
                              hipStream_t stream) {
  (void)in_sizes; (void)n_in; (void)out_size; (void)ws_size;
  const float* x      = (const float*)d_in[0];
  const float* h0     = (const float*)d_in[1];
  const float* c0     = (const float*)d_in[2];
  const float* w_ih0  = (const float*)d_in[3];
  const float* w_hh0  = (const float*)d_in[4];
  const float* b_ih0  = (const float*)d_in[5];
  const float* b_hh0  = (const float*)d_in[6];
  const float* w_ih1  = (const float*)d_in[7];
  const float* w_hh1  = (const float*)d_in[8];
  const float* b_ih1  = (const float*)d_in[9];
  const float* b_hh1  = (const float*)d_in[10];
  const float* head_w = (const float*)d_in[11];
  const float* head_b = (const float*)d_in[12];
  float* outp = (float*)d_out;

  // Workspace carving (256-B aligned), ~96 MB total.
  uintptr_t cur = (uintptr_t)d_ws;
  auto carve = [&](size_t bytes) -> void* {
    void* p = (void*)cur;
    cur += (bytes + 255) & ~(size_t)255;
    return p;
  };

  const long nX     = (long)B_DIM * T_STEPS * V_DIM;   // 4.19M
  const long nWi0   = (long)4 * H_DIM * V_DIM;         // 1.05M
  const long nWh    = (long)4 * H_DIM * H_DIM;         // 4.19M
  const long nHW    = (long)A_DIM * V_DIM * H_DIM;     // 13.1M
  const long nState = (long)B_DIM * H_DIM;             // 65536

  unsigned short* xb   = (unsigned short*)carve(nX * 2);
  unsigned short* wi0b = (unsigned short*)carve(nWi0 * 2);
  unsigned short* wh0b = (unsigned short*)carve(nWh * 2);
  unsigned short* wi1b = (unsigned short*)carve(nWh * 2);
  unsigned short* wh1b = (unsigned short*)carve(nWh * 2);
  unsigned short* hwb  = (unsigned short*)carve(nHW * 2);
  unsigned short* hs0  = (unsigned short*)carve((long)(T_STEPS + 1) * nState * 2);
  unsigned short* h1b  = (unsigned short*)carve(2 * nState * 2);   // double buffer
  float* c0s = (float*)carve(nState * 4);
  float* c1s = (float*)carve(nState * 4);
  float* h0s = (float*)carve(nState * 4);
  float* h1s = (float*)carve(nState * 4);

  // One-time (per call) bf16 conversions; weights then live in L2 across steps.
  convert_f32_bf16_kernel<<<1024, 256, 0, stream>>>(x, xb, nX);
  convert_f32_bf16_kernel<<<512,  256, 0, stream>>>(w_ih0, wi0b, nWi0);
  convert_f32_bf16_kernel<<<1024, 256, 0, stream>>>(w_hh0, wh0b, nWh);
  convert_f32_bf16_kernel<<<1024, 256, 0, stream>>>(w_ih1, wi1b, nWh);
  convert_f32_bf16_kernel<<<1024, 256, 0, stream>>>(w_hh1, wh1b, nWh);
  convert_f32_bf16_kernel<<<1024, 256, 0, stream>>>(head_w, hwb, nHW);
  convert_f32_bf16_kernel<<<64,   256, 0, stream>>>(h0,          hs0, nState);  // L0 h init
  convert_f32_bf16_kernel<<<64,   256, 0, stream>>>(h0 + nState, h1b, nState);  // L1 h init

  hipMemcpyAsync(h0s, h0,          nState * 4, hipMemcpyDeviceToDevice, stream);
  hipMemcpyAsync(h1s, h0 + nState, nState * 4, hipMemcpyDeviceToDevice, stream);
  hipMemcpyAsync(c0s, c0,          nState * 4, hipMemcpyDeviceToDevice, stream);
  hipMemcpyAsync(c1s, c0 + nState, nState * 4, hipMemcpyDeviceToDevice, stream);

  // Layer 0: x_t is a strided view of xb ([b][t][v], row stride T*V).
  for (int t = 0; t < T_STEPS; ++t) {
    lstm_step_kernel<<<256, 256, 0, stream>>>(
        xb + (long)t * V_DIM, (long)T_STEPS * V_DIM, V_DIM,
        wi0b,
        hs0 + (long)t * nState,          // h_{t-1} (slot 0 = init)
        wh0b,
        b_ih0, b_hh0,
        c0s, h0s,
        hs0 + (long)(t + 1) * nState);   // h_t, also layer-1 input
  }
  // Layer 1: input is hs0 slot t+1; recurrent h double-buffered.
  for (int t = 0; t < T_STEPS; ++t) {
    lstm_step_kernel<<<256, 256, 0, stream>>>(
        hs0 + (long)(t + 1) * nState, (long)H_DIM, H_DIM,
        wi1b,
        h1b + (long)(t & 1) * nState,
        wh1b,
        b_ih1, b_hh1,
        c1s, h1s,
        h1b + (long)((t + 1) & 1) * nState);
  }

  // Heads on final top-layer hidden state (T even -> buffer 0).
  const unsigned short* hlast = h1b + (long)(T_STEPS & 1) * nState;
  head_kernel<<<400, 256, 0, stream>>>(hlast, hwb, head_b, outp);

  // h_n [2,B,H] then c_n [2,B,H] after outs [A,B,V].
  float* hn = outp + (long)A_DIM * B_DIM * V_DIM;
  hipMemcpyAsync(hn,              h0s, nState * 4, hipMemcpyDeviceToDevice, stream);
  hipMemcpyAsync(hn + nState,     h1s, nState * 4, hipMemcpyDeviceToDevice, stream);
  hipMemcpyAsync(hn + 2 * nState, c0s, nState * 4, hipMemcpyDeviceToDevice, stream);
  hipMemcpyAsync(hn + 3 * nState, c1s, nState * 4, hipMemcpyDeviceToDevice, stream);
}
